// SparseAutoencoder_35622458753268
// MI455X (gfx1250) — matrix operand order repaired and checked by reading, not compiled, not run on hardware
//
#include <hip/hip_runtime.h>
#include <hip/hip_bf16.h>
#include <cstdint>

#define D_IN   3584
#define H_DIM  28672
#define N_TOK  4096
#define TOPK   32

#define XHAT_ELEMS  ((size_t)N_TOK * D_IN)       // 14680064
#define Z_ELEMS     ((size_t)N_TOK * H_DIM)      // 117440512
#define REC_OFF     (XHAT_ELEMS + Z_ELEMS)
#define SP_OFF      (REC_OFF + 1)

#define WDECT_BYTES ((size_t)H_DIM * D_IN * 2)   // 205,520,896
#define TIDX_BYTES  ((size_t)N_TOK * TOPK * 4)   // 524,288

typedef __attribute__((ext_vector_type(16))) __bf16 v16bf;
typedef __attribute__((ext_vector_type(8)))  __bf16 v8bf;
typedef __attribute__((ext_vector_type(4)))  __bf16 bf16x4;
typedef __attribute__((ext_vector_type(8)))  float  v8f;

union AFrag { v16bf v; v8bf h[2]; };

// --- async global<->LDS support (guarded: falls back to float4 copies) ------
#if __has_builtin(__builtin_amdgcn_global_load_async_to_lds_b128) && \
    __has_builtin(__builtin_amdgcn_global_store_async_from_lds_b128) && \
    __has_builtin(__builtin_amdgcn_s_wait_asynccnt)
#define HAS_ASYNC_LDS 1
#else
#define HAS_ASYNC_LDS 0
#endif

// Builtin signature (from hipcc diagnostic): (int4 AS1*, int4 AS3*, imm, imm)
typedef int i32x4 __attribute__((ext_vector_type(4)));
typedef i32x4 __attribute__((address_space(1))) g_i32x4;
typedef i32x4 __attribute__((address_space(3))) l_i32x4;

__device__ __forceinline__ g_i32x4* gbl_v4(void* p) {
  return (g_i32x4*)(uintptr_t)p;
}
__device__ __forceinline__ l_i32x4* lds_v4(void* p) {
  // generic shared ptr = {aperture, offset}; low 32 bits are the LDS offset
  return (l_i32x4*)(uint32_t)(uintptr_t)p;
}

// ---------------------------------------------------------------------------
// Kernel 1: W_dec [D_IN][H_DIM] f32  ->  W_decT [H_DIM][D_IN] bf16
// Also zero the two loss accumulators in d_out.
// ---------------------------------------------------------------------------
__global__ __launch_bounds__(256) void k_transpose_wdec(
    const float* __restrict__ Wdec, __bf16* __restrict__ WdecT,
    float* __restrict__ out) {
  __shared__ float tile[32][33];
  const int h0 = blockIdx.x * 32, d0 = blockIdx.y * 32;
  const int tx = threadIdx.x, ty = threadIdx.y;
  if (blockIdx.x == 0 && blockIdx.y == 0 && tx == 0 && ty == 0) {
    out[REC_OFF] = 0.0f;
    out[SP_OFF]  = 0.0f;
  }
  #pragma unroll
  for (int i = ty; i < 32; i += 8)
    tile[i][tx] = Wdec[(size_t)(d0 + i) * H_DIM + h0 + tx];
  __syncthreads();
  #pragma unroll
  for (int i = ty; i < 32; i += 8)
    WdecT[(size_t)(h0 + i) * D_IN + d0 + tx] = (__bf16)tile[tx][i];
}

// ---------------------------------------------------------------------------
// Kernel 2: encoder GEMM  z = relu(x @ W_enc^T + b_enc) -> z region of d_out.
// Block tile 128(M tokens) x 128(N hidden) x K=64; 8 waves, each a 32x64
// sub-tile = 2x4 grid of v_wmma_f32_16x16x32_bf16 accumulators.
// Grid order: token-tile fastest so one 1.8MB W_enc stripe is shared in L2
// by all 32 concurrent M-tiles (x itself is fully L2-resident at 59MB).
// ---------------------------------------------------------------------------
#define BM 128
#define BN 128
#define BK 64
#define LP 72

__global__ __launch_bounds__(256) void k_encoder(
    const float* __restrict__ x, const float* __restrict__ Wenc,
    const float* __restrict__ benc, float* __restrict__ zs) {
  __shared__ __bf16 As[BM * LP];
  __shared__ __bf16 Bs[BN * LP];

  const int tid   = threadIdx.x;
  const int lane  = tid & 31;
  const int wave  = tid >> 5;
  const int waveM = wave & 3;   // 4 slabs of 32 rows
  const int waveN = wave >> 2;  // 2 slabs of 64 cols
  const int gm0 = blockIdx.x * BM;   // token dim (fastest-varying block idx)
  const int gh0 = blockIdx.y * BN;   // hidden dim

  // cooperative tile loaders: 256 threads, 2 threads/row, 32 f32 each
  const int lrow = tid >> 1;
  const int lcol = (tid & 1) * 32;
  const float* arow = x    + (size_t)(gm0 + lrow) * D_IN + lcol;
  const float* brow = Wenc + (size_t)(gh0 + lrow) * D_IN + lcol;

  v8f acc[2][4];
  const v8f vzero = {0.f, 0.f, 0.f, 0.f, 0.f, 0.f, 0.f, 0.f};
  #pragma unroll
  for (int mi = 0; mi < 2; mi++)
    #pragma unroll
    for (int ni = 0; ni < 4; ni++) acc[mi][ni] = vzero;

  const int half = lane >> 4;     // 0: lanes 0-15, 1: lanes 16-31
  const int l15  = lane & 15;
  const int mrow0 = (waveM * 32 + l15) * LP;
  const int nrow0 = (waveN * 64 + l15) * LP;

  for (int k0 = 0; k0 < D_IN; k0 += BK) {
    if (k0 + BK < D_IN) {                      // prefetch next K tile into L2
      __builtin_prefetch(arow + k0 + BK, 0, 3);
      __builtin_prefetch(brow + k0 + BK, 0, 3);
    }
    const float4* a4 = (const float4*)(arow + k0);
    const float4* b4 = (const float4*)(brow + k0);
    #pragma unroll
    for (int i = 0; i < 8; i++) {
      float4 va = a4[i], vb = b4[i];
      bf16x4 pa = {(__bf16)va.x, (__bf16)va.y, (__bf16)va.z, (__bf16)va.w};
      bf16x4 pb = {(__bf16)vb.x, (__bf16)vb.y, (__bf16)vb.z, (__bf16)vb.w};
      *(bf16x4*)(&As[lrow * LP + lcol + i * 4]) = pa;
      *(bf16x4*)(&Bs[lrow * LP + lcol + i * 4]) = pb;
    }
    __syncthreads();

    #pragma unroll
    for (int kk = 0; kk < BK; kk += 32) {
      AFrag a[2], b[4];
      // A fragment (16x32, M x K): lanes 0-15 -> K 0..15, 16-31 -> K 16..31
      const int aoff = kk + half * 16;
      #pragma unroll
      for (int mi = 0; mi < 2; mi++) {
        a[mi].h[0] = *(const v8bf*)(&As[mrow0 + mi * 16 * LP + aoff]);
        a[mi].h[1] = *(const v8bf*)(&As[mrow0 + mi * 16 * LP + aoff + 8]);
      }
      // B fragment (32x16, K x N): lanes 0-15 -> K 0..15, 16-31 -> K 16..31
      const int boff = kk + half * 16;
      #pragma unroll
      for (int ni = 0; ni < 4; ni++) {
        b[ni].h[0] = *(const v8bf*)(&Bs[nrow0 + ni * 16 * LP + boff]);
        b[ni].h[1] = *(const v8bf*)(&Bs[nrow0 + ni * 16 * LP + boff + 8]);
      }
      #pragma unroll
      for (int mi = 0; mi < 2; mi++)
        #pragma unroll
        for (int ni = 0; ni < 4; ni++)
          acc[mi][ni] = __builtin_amdgcn_wmma_f32_16x16x32_bf16(
              false, a[mi].v, false, b[ni].v, (short)0, acc[mi][ni],
              false, false);
    }
    __syncthreads();
  }

  // epilogue: + b_enc, ReLU, store into z region of d_out
  float bv[4];
  #pragma unroll
  for (int ni = 0; ni < 4; ni++)
    bv[ni] = benc[gh0 + waveN * 64 + ni * 16 + l15];
  #pragma unroll
  for (int mi = 0; mi < 2; mi++) {
    #pragma unroll
    for (int ni = 0; ni < 4; ni++) {
      const int col = gh0 + waveN * 64 + ni * 16 + l15;
      #pragma unroll
      for (int r = 0; r < 8; r++) {
        const int row = gm0 + waveM * 32 + mi * 16 + half * 8 + r;
        float v = acc[mi][ni][r] + bv[ni];
        zs[(size_t)row * H_DIM + col] = v > 0.f ? v : 0.f;
      }
    }
  }
}

// ---------------------------------------------------------------------------
// Kernel 3: per-token top-32.  One block per token; the 112KB row is a raw
// copy global->LDS and (after sparsification in LDS) LDS->global: use the
// CDNA5 async LDS DMA path (ASYNCcnt) when available.
// ---------------------------------------------------------------------------
__global__ __launch_bounds__(256) void k_topk(
    float* __restrict__ zs, int* __restrict__ tidx, float* __restrict__ tval,
    float* __restrict__ out) {
  __shared__ float rowv[H_DIM];    // 112 KB, fits in 320 KB WGP LDS
  __shared__ float rmax[256];
  __shared__ int   rpos[256];
  __shared__ float sval[TOPK];
  __shared__ int   sidx[TOPK];
  const int tid = threadIdx.x;
  const int n   = blockIdx.x;
  float* zrow = zs + (size_t)n * H_DIM;

  // ---- load row into LDS (async DMA if available) ----
#if HAS_ASYNC_LDS
  for (int c = tid; c < H_DIM / 4; c += 256)
    __builtin_amdgcn_global_load_async_to_lds_b128(
        gbl_v4(zrow + c * 4), lds_v4(&rowv[c * 4]), 0, 0);
  __builtin_amdgcn_s_wait_asynccnt(0);
#else
  for (int c = tid; c < H_DIM / 4; c += 256)
    *(float4*)(&rowv[c * 4]) = *(const float4*)(zrow + c * 4);
#endif
  __syncthreads();

  // ---- 32 iterative argmax extractions ----
  for (int j = 0; j < TOPK; j++) {
    float best = -1.f; int bi = tid;   // post-ReLU values are >= 0
    for (int i = tid; i < H_DIM; i += 256) {
      float v = rowv[i];
      if (v > best) { best = v; bi = i; }
    }
    rmax[tid] = best; rpos[tid] = bi;
    __syncthreads();
    for (int s = 128; s > 0; s >>= 1) {
      if (tid < s && rmax[tid + s] > rmax[tid]) {
        rmax[tid] = rmax[tid + s]; rpos[tid] = rpos[tid + s];
      }
      __syncthreads();
    }
    if (tid == 0) {
      sval[j] = rmax[0]; sidx[j] = rpos[0];
      rowv[rpos[0]] = -1.f;            // sentinel: never re-selected
    }
    __syncthreads();
  }

  // ---- sparsify the row in LDS, then bulk-copy it back to global ----
  for (int i = tid; i < H_DIM; i += 256) {
    float v = 0.f;
    #pragma unroll
    for (int j = 0; j < TOPK; j++)
      if (sidx[j] == i) v = sval[j];
    rowv[i] = v;
  }
  __syncthreads();
#if HAS_ASYNC_LDS
  for (int c = tid; c < H_DIM / 4; c += 256)
    __builtin_amdgcn_global_store_async_from_lds_b128(
        gbl_v4(zrow + c * 4), lds_v4(&rowv[c * 4]), 0, 0);
  __builtin_amdgcn_s_wait_asynccnt(0);
#else
  for (int c = tid; c < H_DIM / 4; c += 256)
    *(float4*)(zrow + c * 4) = *(const float4*)(&rowv[c * 4]);
#endif

  if (tid < TOPK) {
    tidx[n * TOPK + tid] = sidx[tid];
    tval[n * TOPK + tid] = sval[tid];
  }
  if (tid == 0) {
    float s = 0.f;
    #pragma unroll
    for (int j = 0; j < TOPK; j++) s += sval[j];   // |v| == v (ReLU)
    atomicAdd(&out[SP_OFF], s * (1.0f / ((float)N_TOK * (float)H_DIM)));
  }
}

// ---------------------------------------------------------------------------
// Kernel 4: sparse decoder + reconstruction loss.  One block per token;
// gather 32 contiguous bf16 rows of W_decT, f32 accumulate.
// ---------------------------------------------------------------------------
__global__ __launch_bounds__(256) void k_decoder(
    const float* __restrict__ x, const float* __restrict__ bdec,
    const __bf16* __restrict__ WdecT, const int* __restrict__ tidx,
    const float* __restrict__ tval, float* __restrict__ out) {
  __shared__ int   sidx[TOPK];
  __shared__ float sval[TOPK];
  __shared__ float red[256];
  const int tid = threadIdx.x;
  const int n   = blockIdx.x;
  if (tid < TOPK) {
    sidx[tid] = tidx[n * TOPK + tid];
    sval[tid] = tval[n * TOPK + tid];
  }
  __syncthreads();

  float acc[D_IN / 256];                       // 14 outputs per thread
  #pragma unroll
  for (int i = 0; i < D_IN / 256; i++) acc[i] = bdec[tid + i * 256];

  for (int j = 0; j < TOPK; j++) {
    const __bf16* wr = WdecT + (size_t)sidx[j] * D_IN;
    const float v = sval[j];
    #pragma unroll
    for (int i = 0; i < D_IN / 256; i++)
      acc[i] = fmaf(v, (float)wr[tid + i * 256], acc[i]);
  }

  float err = 0.f;
  const float* xrow = x + (size_t)n * D_IN;
  float* xhrow = out + (size_t)n * D_IN;
  #pragma unroll
  for (int i = 0; i < D_IN / 256; i++) {
    const int d = tid + i * 256;
    const float xh = acc[i];
    xhrow[d] = xh;
    const float dd = xh - xrow[d];
    err = fmaf(dd, dd, err);
  }
  red[tid] = err;
  __syncthreads();
  for (int s = 128; s > 0; s >>= 1) {
    if (tid < s) red[tid] += red[tid + s];
    __syncthreads();
  }
  if (tid == 0)
    atomicAdd(&out[REC_OFF], red[0] * (1.0f / ((float)N_TOK * (float)D_IN)));
}

// ---------------------------------------------------------------------------
extern "C" void kernel_launch(void* const* d_in, const int* in_sizes, int n_in,
                              void* d_out, int out_size, void* d_ws,
                              size_t ws_size, hipStream_t stream) {
  const float* x    = (const float*)d_in[0];
  const float* Wenc = (const float*)d_in[1];
  const float* benc = (const float*)d_in[2];
  const float* Wdec = (const float*)d_in[3];
  const float* bdec = (const float*)d_in[4];
  float* out = (float*)d_out;

  __bf16* WdecT = (__bf16*)d_ws;                                   // 205.5 MB
  int*    tidx  = (int*)((char*)d_ws + WDECT_BYTES);               // 0.5 MB
  float*  tval  = (float*)((char*)d_ws + WDECT_BYTES + TIDX_BYTES);// 0.5 MB
  float*  zs    = out + XHAT_ELEMS;   // z_sparse region of d_out

  k_transpose_wdec<<<dim3(H_DIM / 32, D_IN / 32), dim3(32, 8), 0, stream>>>(
      Wdec, WdecT, out);
  // token-tile index fastest for L2 reuse of the W_enc stripe
  k_encoder<<<dim3(N_TOK / BM, H_DIM / BN), dim3(256), 0, stream>>>(
      x, Wenc, benc, zs);
  k_topk<<<dim3(N_TOK), dim3(256), 0, stream>>>(zs, tidx, tval, out);
  k_decoder<<<dim3(N_TOK), dim3(256), 0, stream>>>(x, bdec, WdecT, tidx, tval,
                                                   out);
}
